// LinkPredictor_81243601371376
// MI455X (gfx1250) — matrix operand order repaired and checked by reading, not compiled, not run on hardware
//
#include <hip/hip_runtime.h>
#include <hip/hip_bf16.h>

#define NODES 100000
#define EDGES 1600000
#define KCAT  256      // fused K = 128 (agg) + 128 (self)
#define HIDC  128
#define OUTC  64

typedef __attribute__((ext_vector_type(16))) __bf16 v16bf;
typedef __attribute__((ext_vector_type(8)))  __bf16 v8bf;
typedef __attribute__((ext_vector_type(8)))  float  v8f;

// ---------------- utility: zero f32 buffer ----------------
__global__ __launch_bounds__(256) void k_zero_f32(float* __restrict__ p, int n) {
  int i = blockIdx.x * blockDim.x + threadIdx.x;
  if (i < n) p[i] = 0.0f;
}

// ---------------- degree: atomic count per dst ----------------
__global__ __launch_bounds__(256) void k_degree(const int* __restrict__ dst,
                                                float* __restrict__ deg) {
  int e = blockIdx.x * blockDim.x + threadIdx.x;
  if (e < EDGES) atomicAdd(&deg[dst[e]], 1.0f);
}

// ---------------- scatter-sum feat[src] into agg[dst] (128 ch) ----------------
// one thread handles 4 channels of one edge (32 threads / edge)
__global__ __launch_bounds__(256) void k_scatter(const float* __restrict__ feat,
                                                 const int* __restrict__ src,
                                                 const int* __restrict__ dst,
                                                 float* __restrict__ agg) {
  long long t = (long long)blockIdx.x * blockDim.x + threadIdx.x;
  if (t >= (long long)EDGES * 32) return;
  int e    = (int)(t >> 5);
  int part = ((int)t & 31) * 4;
  int s = src[e], d = dst[e];
  const float4 v = *(const float4*)(feat + (size_t)s * 128 + part);
  float* ap = agg + (size_t)d * 128 + part;
  atomicAdd(ap + 0, v.x);
  atomicAdd(ap + 1, v.y);
  atomicAdd(ap + 2, v.z);
  atomicAdd(ap + 3, v.w);
}

// ---------------- build bf16 [mean_agg | self] concat (N x 256) ----------------
__global__ __launch_bounds__(256) void k_prep_cat(const float* __restrict__ agg,
                                                  const float* __restrict__ deg,
                                                  const float* __restrict__ self,
                                                  __bf16* __restrict__ cat) {
  long long t = (long long)blockIdx.x * blockDim.x + threadIdx.x;
  if (t >= (long long)NODES * 256) return;
  int n = (int)(t >> 8);
  int c = (int)t & 255;
  float v;
  if (c < 128) {
    v = agg[(size_t)n * 128 + c] * (1.0f / fmaxf(deg[n], 1.0f));
  } else {
    v = self[(size_t)n * 128 + (c - 128)];
  }
  cat[(size_t)n * 256 + c] = (__bf16)v;
}

// ---------------- convert + concat weights to bf16 [n_out][256] ----------------
__global__ __launch_bounds__(256) void k_prep_w(const float* __restrict__ wl,
                                                const float* __restrict__ wr,
                                                __bf16* __restrict__ wcat, int nrows) {
  int t = blockIdx.x * blockDim.x + threadIdx.x;
  if (t >= nrows * 256) return;
  int n = t >> 8, k = t & 255;
  float v = (k < 128) ? wl[n * 128 + k] : wr[n * 128 + (k - 128)];
  wcat[t] = (__bf16)v;
}

// ---------------- WMMA A-fragment loader ----------------
// A 16x32 bf16: lane l holds row (l&15); K chunk [kb..kb+15], kb=(l>>4)*16
__device__ inline v16bf load_a_frag(const __bf16* p) {
  v8bf lo = *(const v8bf*)(p);
  v8bf hi = *(const v8bf*)(p + 8);
  v16bf f;
#pragma unroll
  for (int j = 0; j < 8; ++j) { f[j] = lo[j]; f[j + 8] = hi[j]; }
  return f;
}

// ---------------- fused GEMM: out[N][n_out] = cat @ wcat^T + bias (opt. ReLU) ----
// Block = 8 waves; each block owns 8 M-tiles x one 64-column group.
// Weight tile (64 x 256 bf16 = 32 KB) staged in LDS once per block; each wave
// computes a 16(M) x 64(N) tile with K = 256 in 8 steps of 32.
__global__ __launch_bounds__(256)
void k_wmma_gemm(const __bf16* __restrict__ in, const __bf16* __restrict__ w,
                 const float* __restrict__ bias, float* __restrict__ out,
                 int n_out, int do_relu, int mtiles) {
  __shared__ __bf16 wlds[64 * KCAT];            // 32 KB of the 320 KB / WGP

  int ngrp = n_out >> 6;                        // 64-column groups (1 or 2)
  int ng   = (int)(blockIdx.x) % ngrp;
  int mblk = (int)(blockIdx.x) / ngrp;
  int warp = threadIdx.x >> 5;
  int lane = threadIdx.x & 31;

  // ---- cooperative stage of this block's weight tile (contiguous 32 KB) ----
  {
    const __bf16* wsrc = w + (size_t)(ng * 64) * KCAT;
    int t = threadIdx.x;
#pragma unroll
    for (int i = 0; i < 8; ++i) {               // 256 thr * 8 it * 8 elem = 16384
      int idx = (t + i * 256) * 8;
      *(v8bf*)(wlds + idx) = *(const v8bf*)(wsrc + idx);
    }
  }
  __syncthreads();

  int mtile = mblk * 8 + warp;
  if (mtile >= mtiles) return;                  // wave-uniform: EXEC all-1s below

  int lhalf = lane >> 4;                        // 0 or 1
  int l15   = lane & 15;
  const __bf16* arow = in + (size_t)(mtile * 16 + l15) * KCAT;
  // B fragments come from LDS: lane l holds column (l&15)+nt*16,
  // K = contiguous chunk [(l>>4)*16 .. +15] per 32-K step
  const __bf16* bp0 = wlds + (size_t)(l15)      * KCAT;
  const __bf16* bp1 = wlds + (size_t)(l15 + 16) * KCAT;
  const __bf16* bp2 = wlds + (size_t)(l15 + 32) * KCAT;
  const __bf16* bp3 = wlds + (size_t)(l15 + 48) * KCAT;

  int akb = lhalf * 16;   // A: contiguous 16-K chunk per lane-half
  int bkb = lhalf * 16;   // B: contiguous 16-K chunk per lane-half

  v8f acc0 = {}, acc1 = {}, acc2 = {}, acc3 = {};
#pragma unroll
  for (int kk = 0; kk < KCAT / 32; ++kk) {
    int k0 = kk * 32;
    v16bf a  = load_a_frag(arow + k0 + akb);
    v16bf b0 = *(const v16bf*)(bp0 + k0 + bkb);
    v16bf b1 = *(const v16bf*)(bp1 + k0 + bkb);
    v16bf b2 = *(const v16bf*)(bp2 + k0 + bkb);
    v16bf b3 = *(const v16bf*)(bp3 + k0 + bkb);
    acc0 = __builtin_amdgcn_wmma_f32_16x16x32_bf16(false, a, false, b0, (short)0, acc0, false, false);
    acc1 = __builtin_amdgcn_wmma_f32_16x16x32_bf16(false, a, false, b1, (short)0, acc1, false, false);
    acc2 = __builtin_amdgcn_wmma_f32_16x16x32_bf16(false, a, false, b2, (short)0, acc2, false, false);
    acc3 = __builtin_amdgcn_wmma_f32_16x16x32_bf16(false, a, false, b3, (short)0, acc3, false, false);
  }

  // C/D layout: lane l -> N = l&15 ; VGPR r -> M = r + 8*(l>>4)
  float bv0 = bias[ng * 64 + l15];
  float bv1 = bias[ng * 64 + l15 + 16];
  float bv2 = bias[ng * 64 + l15 + 32];
  float bv3 = bias[ng * 64 + l15 + 48];
  float* orow = out + (size_t)(mtile * 16 + lhalf * 8) * n_out + ng * 64 + l15;
#pragma unroll
  for (int r = 0; r < 8; ++r) {
    float v0 = acc0[r] + bv0;
    float v1 = acc1[r] + bv1;
    float v2 = acc2[r] + bv2;
    float v3 = acc3[r] + bv3;
    if (do_relu) {
      v0 = fmaxf(v0, 0.0f); v1 = fmaxf(v1, 0.0f);
      v2 = fmaxf(v2, 0.0f); v3 = fmaxf(v3, 0.0f);
    }
    float* po = orow + (size_t)r * n_out;
    po[0] = v0; po[16] = v1; po[32] = v2; po[48] = v3;
  }
}

// ---------------- decode: wave per edge, dot over 64 channels ----------------
__global__ __launch_bounds__(256) void k_decode(const float* __restrict__ z,
                                                const int* __restrict__ src,
                                                const int* __restrict__ dst,
                                                float* __restrict__ out) {
  long long t = (long long)blockIdx.x * blockDim.x + threadIdx.x;
  if (t >= (long long)EDGES * 32) return;
  int e    = (int)(t >> 5);
  int lane = (int)t & 31;
  int s = src[e], d = dst[e];
  const float* zs = z + (size_t)s * 64;
  const float* zd = z + (size_t)d * 64;
  float p = zs[lane] * zd[lane] + zs[lane + 32] * zd[lane + 32];
#pragma unroll
  for (int off = 16; off > 0; off >>= 1) p += __shfl_xor(p, off, 32);
  if (lane == 0) out[e] = p;
}

// ---------------- host launcher ----------------
extern "C" void kernel_launch(void* const* d_in, const int* in_sizes, int n_in,
                              void* d_out, int out_size, void* d_ws, size_t ws_size,
                              hipStream_t stream) {
  (void)in_sizes; (void)n_in; (void)out_size; (void)ws_size;
  const float* x   = (const float*)d_in[0];
  const int*   ei  = (const int*)d_in[1];
  const float* w1l = (const float*)d_in[2];
  const float* w1r = (const float*)d_in[3];
  const float* b1  = (const float*)d_in[4];
  const float* w2l = (const float*)d_in[5];
  const float* w2r = (const float*)d_in[6];
  const float* b2  = (const float*)d_in[7];
  float* out = (float*)d_out;
  const int* src = ei;
  const int* dst = ei + EDGES;

  char* ws = (char*)d_ws;
  size_t off = 0;
  auto alloc = [&](size_t bytes) -> void* {
    void* p = ws + off;
    off = (off + bytes + 255) & ~(size_t)255;
    return p;
  };
  float*  deg  = (float*) alloc((size_t)NODES * 4);            // degrees
  float*  aggF = (float*) alloc((size_t)NODES * 128 * 4);      // scatter accum (reused)
  __bf16* catB = (__bf16*)alloc((size_t)NODES * 256 * 2);      // bf16 concat (reused)
  float*  hF   = (float*) alloc((size_t)NODES * 128 * 4);      // layer-1 output
  float*  zF   = (float*) alloc((size_t)NODES * 64 * 4);       // layer-2 output
  __bf16* w1c  = (__bf16*)alloc((size_t)128 * 256 * 2);
  __bf16* w2c  = (__bf16*)alloc((size_t)64 * 256 * 2);

  const int mt   = NODES / 16;                 // 6250 exact
  const int mblk = (mt + 7) / 8;               // 8 M-tiles per block
  long long eth = (long long)EDGES * 32;       // edge threads (scatter/decode)
  long long nth = (long long)NODES * 256;      // concat threads

  // weights -> bf16
  k_prep_w<<<(128 * 256 + 255) / 256, 256, 0, stream>>>(w1l, w1r, w1c, 128);
  k_prep_w<<<(64 * 256 + 255) / 256, 256, 0, stream>>>(w2l, w2r, w2c, 64);

  // degree
  k_zero_f32<<<(NODES + 255) / 256, 256, 0, stream>>>(deg, NODES);
  k_degree<<<(EDGES + 255) / 256, 256, 0, stream>>>(dst, deg);

  // ---- layer 1 ----
  k_zero_f32<<<(NODES * 128 + 255) / 256, 256, 0, stream>>>(aggF, NODES * 128);
  k_scatter<<<(int)((eth + 255) / 256), 256, 0, stream>>>(x, src, dst, aggF);
  k_prep_cat<<<(int)((nth + 255) / 256), 256, 0, stream>>>(aggF, deg, x, catB);
  k_wmma_gemm<<<mblk * (HIDC / 64), 256, 0, stream>>>(catB, w1c, b1, hF, HIDC, 1, mt);

  // ---- layer 2 ----
  k_zero_f32<<<(NODES * 128 + 255) / 256, 256, 0, stream>>>(aggF, NODES * 128);
  k_scatter<<<(int)((eth + 255) / 256), 256, 0, stream>>>(hF, src, dst, aggF);
  k_prep_cat<<<(int)((nth + 255) / 256), 256, 0, stream>>>(aggF, deg, hF, catB);
  k_wmma_gemm<<<mblk * (OUTC / 64), 256, 0, stream>>>(catB, w2c, b2, zF, OUTC, 0, mt);

  // ---- decode ----
  k_decode<<<(int)((eth + 255) / 256), 256, 0, stream>>>(zF, src, dst, out);
}
